// OptimizedFiconnModel_22376779612869
// MI455X (gfx1250) — matrix operand order repaired and checked by reading, not compiled, not run on hardware
//
#include <hip/hip_runtime.h>
#include <hip/hip_bf16.h>
#include <stdint.h>

typedef __attribute__((ext_vector_type(16))) _Float16 v16h;
typedef __attribute__((ext_vector_type(8)))  _Float16 v8h;
typedef __attribute__((ext_vector_type(4)))  _Float16 v4h;
typedef __attribute__((ext_vector_type(8)))  float    v8f;
typedef __attribute__((ext_vector_type(4)))  float    v4f;
typedef __attribute__((ext_vector_type(4)))  uint32_t v4u;
typedef __attribute__((ext_vector_type(8)))  int      v8i;
typedef __attribute__((ext_vector_type(4)))  int      v4i;

#define NN      64
#define CLSN    12
#define BATCH   262144
#define GAMMA_F 0.05f
#define G2_F    0.0025f
#define CH      0.70710678118654752f

// workspace half-region layout (offsets in halves)
#define WS_U      0        // 5 layers * 8192 (Ur[64*64] then Ui[64*64])
#define WS_W1     40960    // 8192  (128x64 row-major)
#define WS_W2     49152    // 8192  (64x128 row-major)
#define WS_W3     57344    // 1024  (16x64, rows >=12 zero)
#define WS_HALVES 58368
// float region after halves: amp[256], bc[256]

#if defined(__has_builtin)
#if __has_builtin(__builtin_amdgcn_tensor_load_to_lds) && \
    __has_builtin(__builtin_amdgcn_s_wait_tensorcnt)
#define HAVE_TDM 1
#endif
#endif

static __device__ __forceinline__ v8f zero8() {
  v8f c;
#pragma unroll
  for (int i = 0; i < 8; i++) c[i] = 0.0f;
  return c;
}

static __device__ __forceinline__ v16h vneg16(v16h a) {
  v16h r;
#pragma unroll
  for (int i = 0; i < 16; i++) r[i] = -a[i];
  return r;
}

static __device__ __forceinline__ v8f wmma16(v16h a, v16h b, v8f c) {
  // D = A(16x32 f16) x B(32x16 f16) + C(16x16 f32)
  return __builtin_amdgcn_wmma_f32_16x16x32_f16(
      false, a, false, b, (short)0, c, false, false);
}

// A-fragment (16x32 f16) from row-major [16][stride] LDS tile.
static __device__ __forceinline__ v16h ldsA(const _Float16* base, int stride,
                                            int lane, int kc) {
  int row = lane & 15;
  int kb  = ((lane >> 4) << 3) + (kc << 5);
  const _Float16* p = base + row * stride + kb;
  v8h lo = *(const v8h*)p;
  v8h hi = *(const v8h*)(p + 16);
  return __builtin_shufflevector(lo, hi, 0, 1, 2, 3, 4, 5, 6, 7,
                                 8, 9, 10, 11, 12, 13, 14, 15);
}

// B-fragment (32x16 f16): B[k][n] = M[n][k] for row-major weight M.
static __device__ __forceinline__ v16h ldsB(const _Float16* rowmajor, int stride,
                                            int lane, int nt, int kc) {
  int row = (nt << 4) + (lane & 15);
  int kb  = (kc << 5) + ((lane >> 4) << 3);
  const _Float16* p = rowmajor + row * stride + kb;
  v8h lo = *(const v8h*)p;
  v8h hi = *(const v8h*)(p + 16);
  return __builtin_shufflevector(lo, hi, 0, 1, 2, 3, 4, 5, 6, 7,
                                 8, 9, 10, 11, 12, 13, 14, 15);
}

// ---------------------------------------------------------------------------
// Prep: build the 5 complex unitaries (MZI mesh scan), convert weights to f16,
// precompute nofu coefficients. Tiny kernel, 8 blocks x 64 threads.
// ---------------------------------------------------------------------------
__global__ __launch_bounds__(64) void ficonn_prep(
    const float* __restrict__ mzi, const float* __restrict__ oph,
    const float* __restrict__ betap, const float* __restrict__ W1,
    const float* __restrict__ W2, const float* __restrict__ W3,
    _Float16* __restrict__ wsh, float* __restrict__ wsf) {
  const int b = blockIdx.x;
  const int t = threadIdx.x;
  if (b < 5) {
    __shared__ float Ure[64][64];
    __shared__ float Uim[64][64];
    __shared__ float ph[4032];
    for (int i = t; i < 4032; i += 64) ph[i] = mzi[b * 4032 + i];
    for (int i = 0; i < 64; i++) {
      Ure[i][t] = (i == t) ? 1.0f : 0.0f;
      Uim[i][t] = 0.0f;
    }
    __syncthreads();
    // thread t owns column t; the MZI scan mixes rows only -> no barriers.
    int k = 0;
    for (int i = 0; i < 63; i++) {
      for (int j = i + 1; j < 64; j++, k++) {
        float th = ph[2 * k], phi = ph[2 * k + 1];
        float st, ct, sp, cp;
        __sincosf(th, &st, &ct);
        __sincosf(phi, &sp, &cp);
        float ar = Ure[i][t], ai = Uim[i][t];
        float br = Ure[j][t], bi = Uim[j][t];
        float er = cp * br - sp * bi;   // e^{i phi} * u_j
        float ei = cp * bi + sp * br;
        float dr = ar - er, di = ai - ei;
        Ure[i][t] = CH * (ar + er);
        Uim[i][t] = CH * (ai + ei);
        Ure[j][t] = CH * (ct * dr - st * di);
        Uim[j][t] = CH * (ct * di + st * dr);
      }
    }
    for (int i = 0; i < 64; i++) {
      float o = oph[b * 64 + i];
      float so, co;
      __sincosf(o, &so, &co);
      float rr = co * Ure[i][t] - so * Uim[i][t];
      float ii = co * Uim[i][t] + so * Ure[i][t];
      wsh[WS_U + b * 8192 + i * 64 + t]        = (_Float16)rr;
      wsh[WS_U + b * 8192 + 4096 + i * 64 + t] = (_Float16)ii;
    }
  } else if (b == 5) {
    for (int i = t; i < 8192; i += 64) wsh[WS_W1 + i] = (_Float16)W1[i];
  } else if (b == 6) {
    for (int i = t; i < 8192; i += 64) wsh[WS_W2 + i] = (_Float16)W2[i];
  } else {
    for (int i = t; i < 1024; i += 64) {
      int o = i >> 6, j = i & 63;
      wsh[WS_W3 + i] = (o < CLSN) ? (_Float16)W3[o * 64 + j] : (_Float16)0.0f;
    }
    for (int i = t; i < 256; i += 64) {
      float s = 1.0f / (1.0f + __expf(-betap[i]));
      wsf[i]       = sqrtf(fmaxf(0.0f, 1.0f - s));  // amplitude
      wsf[256 + i] = 0.0008f * s;                   // SENS*RESP*beta
    }
  }
}

// ---------------------------------------------------------------------------
// Main fused kernel: 128 batch rows per workgroup (8 waves x 16 rows).
// U's + MLP weights pulled into LDS by the Tensor Data Mover; per-wave
// private staging buffer for layout conversion between WMMA C/D and A frags.
// ---------------------------------------------------------------------------
__global__ __launch_bounds__(256) void ficonn_main(
    const float* __restrict__ x, const float* __restrict__ det0,
    const float* __restrict__ b1, const float* __restrict__ b2,
    const float* __restrict__ b3, const _Float16* __restrict__ wsh,
    const float* __restrict__ wsf, float* __restrict__ out) {
  __shared__ __align__(16) _Float16 sH[WS_HALVES];   // U's + W1 + W2 + W3
  __shared__ __align__(16) _Float16 sStage[8][2048]; // per-wave 16x128 staging
  __shared__ float sAmp[256], sBc[256], sD0[256], sB1[128], sB2[64];

  const int tid  = threadIdx.x;
  const int lane = tid & 31;
  const int w    = tid >> 5;

  // Prefetch this block's 32KB x tile while the LDS fill is in flight.
  {
    const float* xb = x + (size_t)blockIdx.x * 128 * 64;
    __builtin_prefetch(xb + (size_t)tid * 32, 0, 0);
  }

#ifdef HAVE_TDM
  if (w == 0) {
    // 1-D TDM copy: WS_HALVES*2 bytes as 14592 x 8-byte elements.
    const uint32_t nelem = (WS_HALVES * 2) / 8;
    const uint32_t ldsa  = (uint32_t)(uintptr_t)(void*)&sH[0];
    const uint64_t ga    = (uint64_t)(uintptr_t)(const void*)wsh;
    v4u g0;
    g0[0] = 1u;                                  // count=1, user mode
    g0[1] = ldsa;                                // lds_addr
    g0[2] = (uint32_t)ga;                        // global_addr[31:0]
    g0[3] = (uint32_t)((ga >> 32) & 0x1FFFFFFu) | (2u << 30);  // addr hi | type=2
    v8i g1;
    g1[0] = (3 << 16);                           // workgroup_mask=0, data_size=8B
    g1[1] = (int)((nelem & 0xFFFFu) << 16);      // tensor_dim0[15:0]
    g1[2] = (int)((nelem >> 16) | (1u << 16));   // tensor_dim0[31:16], tensor_dim1=1
    g1[3] = (int)((nelem & 0xFFFFu) << 16);      // tile_dim0 = nelem
    g1[4] = 0;                                   // tile_dim1/2 = 0 (1-D)
    g1[5] = (int)nelem;                          // tensor_dim0_stride
    g1[6] = 0;
    g1[7] = 0;
    v4i z4;
    z4[0] = 0; z4[1] = 0; z4[2] = 0; z4[3] = 0;
    v8i z8;
#pragma unroll
    for (int q = 0; q < 8; q++) z8[q] = 0;
    __builtin_amdgcn_tensor_load_to_lds(g0, g1, z4, z4, z8, 0);
    __builtin_amdgcn_s_wait_tensorcnt(0);
  }
#else
  for (int i = tid; i < WS_HALVES / 8; i += 256)
    ((v8h*)sH)[i] = ((const v8h*)wsh)[i];
#endif
  if (tid < 256) {
    sAmp[tid] = wsf[tid];
    sBc[tid]  = wsf[256 + tid];
    sD0[tid]  = det0[tid];
  }
  if (tid < 128) sB1[tid] = b1[tid];
  if (tid < 64)  sB2[tid] = b2[tid];
  __syncthreads();

  const int rowBase = blockIdx.x * 128 + w * 16;
  _Float16* stg = &sStage[w][0];
  _Float16* re  = stg;        // [16][64]
  _Float16* im  = stg + 1024; // [16][64]

  // ---- load x tile (f32, coalesced b128) -> f16 re (field is purely real) ----
#pragma unroll
  for (int i = 0; i < 8; i++) {
    int idx4 = i * 32 + lane;
    int r    = idx4 >> 4;
    int c4   = (idx4 & 15) << 2;
    v4f xv = *reinterpret_cast<const v4f*>(x + (size_t)(rowBase + r) * 64 + c4);
    v4h hv;
#pragma unroll
    for (int q = 0; q < 4; q++) hv[q] = (_Float16)xv[q];
    *(v4h*)(re + r * 64 + c4) = hv;
  }

  const int n0 = lane & 15;
  const int mb = (lane >> 4) << 3;

  // ---- layer 0 (imaginary part of field is zero: half the WMMAs) ----
  {
    const _Float16* Ur = sH + WS_U;
    const _Float16* Ui = Ur + 4096;
    v16h frA[2];
    frA[0] = ldsA(re, 64, lane, 0);
    frA[1] = ldsA(re, 64, lane, 1);
#pragma unroll
    for (int t = 0; t < 4; t++) {
      v8f cr = zero8(), ci = zero8();
#pragma unroll
      for (int kc = 0; kc < 2; kc++) {
        cr = wmma16(frA[kc], ldsB(Ur, 64, lane, t, kc), cr);  // Re = fr*Ur
        ci = wmma16(frA[kc], ldsB(Ui, 64, lane, t, kc), ci);  // Im = fr*Ui
      }
      int n = (t << 4) + n0;
      float amp = sAmp[n], bc = sBc[n], dd0 = sD0[n];
#pragma unroll
      for (int r = 0; r < 8; r++) {
        float rr = cr[r], ii = ci[r];
        float P  = rr * rr + ii * ii;
        float d  = dd0 + bc * P;
        float q  = 1.0f / (G2_F + d * d);
        float tr = G2_F * q;
        float ti = -GAMMA_F * d * q;
        re[(mb + r) * 64 + n] = (_Float16)(amp * (rr * tr - ii * ti));
        im[(mb + r) * 64 + n] = (_Float16)(amp * (rr * ti + ii * tr));
      }
    }
  }

  // ---- layers 1..4: full complex matvec (+ nofu, or |.|^2 on the last) ----
#pragma unroll
  for (int l = 1; l < 5; l++) {
    const _Float16* Ur = sH + WS_U + l * 8192;
    const _Float16* Ui = Ur + 4096;
    v16h frA[2], fiA[2], fnA[2];
#pragma unroll
    for (int kc = 0; kc < 2; kc++) {
      frA[kc] = ldsA(re, 64, lane, kc);
      fiA[kc] = ldsA(im, 64, lane, kc);
      fnA[kc] = vneg16(fiA[kc]);
    }
    v8f accR[4], accI[4];
#pragma unroll
    for (int t = 0; t < 4; t++) {
      v8f cr = zero8(), ci = zero8();
#pragma unroll
      for (int kc = 0; kc < 2; kc++) {
        v16h br = ldsB(Ur, 64, lane, t, kc);
        v16h bi = ldsB(Ui, 64, lane, t, kc);
        cr = wmma16(frA[kc], br, cr);  // Re += fr*Ur
        cr = wmma16(fnA[kc], bi, cr);  // Re -= fi*Ui
        ci = wmma16(fiA[kc], br, ci);  // Im += fi*Ur
        ci = wmma16(frA[kc], bi, ci);  // Im += fr*Ui
      }
      accR[t] = cr;
      accI[t] = ci;
    }
    if (l < 4) {
#pragma unroll
      for (int t = 0; t < 4; t++) {
        int n = (t << 4) + n0;
        float amp = sAmp[l * 64 + n], bc = sBc[l * 64 + n], dd0 = sD0[l * 64 + n];
#pragma unroll
        for (int r = 0; r < 8; r++) {
          float rr = accR[t][r], ii = accI[t][r];
          float P  = rr * rr + ii * ii;
          float d  = dd0 + bc * P;
          float q  = 1.0f / (G2_F + d * d);
          float tr = G2_F * q;
          float ti = -GAMMA_F * d * q;
          re[(mb + r) * 64 + n] = (_Float16)(amp * (rr * tr - ii * ti));
          im[(mb + r) * 64 + n] = (_Float16)(amp * (rr * ti + ii * tr));
        }
      }
    } else {
      // last layer: stage |field|^2 for the MLP
#pragma unroll
      for (int t = 0; t < 4; t++) {
        int n = (t << 4) + n0;
#pragma unroll
        for (int r = 0; r < 8; r++) {
          float rr = accR[t][r], ii = accI[t][r];
          re[(mb + r) * 64 + n] = (_Float16)(rr * rr + ii * ii);
        }
      }
    }
  }

  // ---- MLP layer 1: 64 -> 128, relu ----
  {
    v16h pA[2];
    pA[0] = ldsA(re, 64, lane, 0);
    pA[1] = ldsA(re, 64, lane, 1);
    v8f h1[8];
#pragma unroll
    for (int t = 0; t < 8; t++) {
      v8f c = zero8();
#pragma unroll
      for (int kc = 0; kc < 2; kc++)
        c = wmma16(pA[kc], ldsB(sH + WS_W1, 64, lane, t, kc), c);
      float bias = sB1[(t << 4) + n0];
#pragma unroll
      for (int r = 0; r < 8; r++) c[r] = fmaxf(c[r] + bias, 0.0f);
      h1[t] = c;
    }
#pragma unroll
    for (int t = 0; t < 8; t++) {
      int n = (t << 4) + n0;
#pragma unroll
      for (int r = 0; r < 8; r++) stg[(mb + r) * 128 + n] = (_Float16)h1[t][r];
    }
  }

  // ---- MLP layer 2: 128 -> 64, relu ----
  {
    v16h aA[4];
#pragma unroll
    for (int kc = 0; kc < 4; kc++) aA[kc] = ldsA(stg, 128, lane, kc);
    v8f h2[4];
#pragma unroll
    for (int t = 0; t < 4; t++) {
      v8f c = zero8();
#pragma unroll
      for (int kc = 0; kc < 4; kc++)
        c = wmma16(aA[kc], ldsB(sH + WS_W2, 128, lane, t, kc), c);
      float bias = sB2[(t << 4) + n0];
#pragma unroll
      for (int r = 0; r < 8; r++) c[r] = fmaxf(c[r] + bias, 0.0f);
      h2[t] = c;
    }
#pragma unroll
    for (int t = 0; t < 4; t++) {
      int n = (t << 4) + n0;
#pragma unroll
      for (int r = 0; r < 8; r++) stg[(mb + r) * 64 + n] = (_Float16)h2[t][r];
    }
  }

  // ---- MLP layer 3: 64 -> 12 (padded to 16), write out ----
  {
    v8f c = zero8();
#pragma unroll
    for (int kc = 0; kc < 2; kc++)
      c = wmma16(ldsA(stg, 64, lane, kc), ldsB(sH + WS_W3, 64, lane, 0, kc), c);
    if (n0 < CLSN) {
      float bias = b3[n0];
#pragma unroll
      for (int r = 0; r < 8; r++)
        out[(size_t)(rowBase + mb + r) * CLSN + n0] = c[r] + bias;
    }
  }
}

extern "C" void kernel_launch(void* const* d_in, const int* in_sizes, int n_in,
                              void* d_out, int out_size, void* d_ws,
                              size_t ws_size, hipStream_t stream) {
  const float* x    = (const float*)d_in[0];
  const float* mzi  = (const float*)d_in[1];
  const float* oph  = (const float*)d_in[2];
  const float* beta = (const float*)d_in[3];
  const float* det  = (const float*)d_in[4];
  const float* W1   = (const float*)d_in[5];
  const float* b1   = (const float*)d_in[6];
  const float* W2   = (const float*)d_in[7];
  const float* b2   = (const float*)d_in[8];
  const float* W3   = (const float*)d_in[9];
  const float* b3   = (const float*)d_in[10];

  _Float16* wsh = (_Float16*)d_ws;
  float* wsf = (float*)((char*)d_ws + WS_HALVES * sizeof(_Float16));

  ficonn_prep<<<8, 64, 0, stream>>>(mzi, oph, beta, W1, W2, W3, wsh, wsf);
  ficonn_main<<<BATCH / 128, 256, 0, stream>>>(x, det, b1, b2, b3, wsh, wsf,
                                               (float*)d_out);
}
